// DotDecoder_32607391711805
// MI455X (gfx1250) — matrix operand order repaired and checked by reading, not compiled, not run on hardware
//
#include <hip/hip_runtime.h>

// DotDecoder: out[e] = sigmoid( dot(c_feat[src[e]], g_feat[dst[e]]) ), D=64.
// Gather-bound: feature tables (~31 MB) are L2-resident on MI455X (192 MB L2);
// 16 edges/wave batched as diagonal of a 16x16 f16 WMMA with f32 accumulate.
// Sigmoid uses v_exp_f32 + v_rcp_f32 (approximate reciprocal) instead of the
// IEEE division expansion.

typedef __attribute__((ext_vector_type(16))) _Float16 v16h;
typedef __attribute__((ext_vector_type(8)))  float    v8f;
typedef __attribute__((ext_vector_type(4)))  float    f32x4;

__device__ __forceinline__ f32x4 ld4(const float* p) {
    return *reinterpret_cast<const f32x4*>(p);
}

__device__ __forceinline__ float fast_sigmoid(float x) {
    // 1 / (1 + e^-x) with hardware v_exp_f32 / v_rcp_f32
    return __builtin_amdgcn_rcpf(1.0f + __expf(-x));
}

// Convert 8 consecutive floats (two f32x4) into fragment halves [i..i+7].
__device__ __forceinline__ void pack8(v16h& d, int i, f32x4 lo, f32x4 hi) {
    d[i + 0] = (_Float16)lo[0]; d[i + 1] = (_Float16)lo[1];
    d[i + 2] = (_Float16)lo[2]; d[i + 3] = (_Float16)lo[3];
    d[i + 4] = (_Float16)hi[0]; d[i + 5] = (_Float16)hi[1];
    d[i + 6] = (_Float16)hi[2]; d[i + 7] = (_Float16)hi[3];
}

__global__ __launch_bounds__(256) void DotDecoder_wmma_kernel(
    const float* __restrict__ c_feat,   // [N_CELLS, 64]
    const float* __restrict__ g_feat,   // [N_GENES, 64]
    const int*   __restrict__ src_idx,  // [E]
    const int*   __restrict__ dst_idx,  // [E]
    float*       __restrict__ out,      // [E]
    int nEdges)
{
    const int lane = threadIdx.x & 31;
    const int wave = threadIdx.x >> 5;
    const int gid  = blockIdx.x * 8 + wave;   // 16-edge group id (wave-uniform)
    const int fullGroups = nEdges >> 4;
    const int tail       = nEdges & 15;

    if (gid < fullGroups) {
        const int h  = lane >> 4;   // which 16-lane half
        const int m  = lane & 15;   // edge-in-group == matrix row/col
        const int e0 = gid << 4;

        // Indices streamed once: non-temporal so they don't evict feature rows.
        const int srow = __builtin_nontemporal_load(src_idx + e0 + m);
        const int drow = __builtin_nontemporal_load(dst_idx + e0 + m);
        const float* crow = c_feat + (size_t)srow * 64;
        const float* grow = g_feat + (size_t)drow * 64;

        v8f acc = {};
#pragma unroll
        for (int kb = 0; kb < 2; ++kb) {          // K = 64 as 2 x 32
            const float* ca = crow + kb * 32;
            const float* gb = grow + kb * 32;
            // A fragment (16-bit 16x32 layout): lane holds K = 8h..8h+7 and 16+8h..16+8h+7
            f32x4 a0 = ld4(ca + 8 * h);
            f32x4 a1 = ld4(ca + 8 * h + 4);
            f32x4 a2 = ld4(ca + 16 + 8 * h);
            f32x4 a3 = ld4(ca + 16 + 8 * h + 4);
            // B fragment (32x16 layout): lane holds K = 8h..8h+7 and 16+8h..16+8h+7
            f32x4 b0 = ld4(gb + 8 * h);
            f32x4 b1 = ld4(gb + 8 * h + 4);
            f32x4 b2 = ld4(gb + 16 + 8 * h);
            f32x4 b3 = ld4(gb + 16 + 8 * h + 4);

            v16h A, B;
            pack8(A, 0, a0, a1);
            pack8(A, 8, a2, a3);
            pack8(B, 0, b0, b1);
            pack8(B, 8, b2, b3);

            acc = __builtin_amdgcn_wmma_f32_16x16x32_f16(
                /*neg_a=*/false, A, /*neg_b=*/false, B,
                /*c_mod=*/(short)0, acc, /*reuse_a=*/false, /*reuse_b=*/false);
        }

        // Diagonal D[m][m]: lane m holds acc[m] (m<8); lane m+16 holds acc[m-8] (m>=8).
        const int r = lane & 7;
        float t01 = (r & 1) ? acc[1] : acc[0];
        float t23 = (r & 1) ? acc[3] : acc[2];
        float t45 = (r & 1) ? acc[5] : acc[4];
        float t67 = (r & 1) ? acc[7] : acc[6];
        float t03 = (r & 2) ? t23 : t01;
        float t47 = (r & 2) ? t67 : t45;
        float diag = (r & 4) ? t47 : t03;

        // Route score of edge m to lane m (ds_bpermute).
        const int srcLane = ((lane & 8) ? (lane + 16) : lane) & 31;
        float score = __shfl(diag, srcLane, 32);

        if (lane < 16) {
            __builtin_nontemporal_store(fast_sigmoid(score), out + e0 + lane);
        }
    } else if (tail && gid == fullGroups) {
        // Scalar tail (not taken for E % 16 == 0, kept for generality).
        const int e = (fullGroups << 4) + lane;
        if (lane < tail) {
            const float* crow = c_feat + (size_t)__builtin_nontemporal_load(src_idx + e) * 64;
            const float* grow = g_feat + (size_t)__builtin_nontemporal_load(dst_idx + e) * 64;
            float s = 0.0f;
#pragma unroll
            for (int k = 0; k < 64; ++k) s += crow[k] * grow[k];
            __builtin_nontemporal_store(fast_sigmoid(s), out + e);
        }
    }
}

extern "C" void kernel_launch(void* const* d_in, const int* in_sizes, int n_in,
                              void* d_out, int out_size, void* d_ws, size_t ws_size,
                              hipStream_t stream) {
    const float* c_feat  = (const float*)d_in[0];
    const float* g_feat  = (const float*)d_in[1];
    const int*   src_idx = (const int*)d_in[2];
    const int*   dst_idx = (const int*)d_in[3];
    float*       out     = (float*)d_out;

    const int E      = in_sizes[2];            // number of edges
    const int groups = (E + 15) / 16;          // 16 edges per wave
    const int wavesPerBlock = 8;               // 256 threads, wave32
    const int blocks = (groups + wavesPerBlock - 1) / wavesPerBlock;

    DotDecoder_wmma_kernel<<<blocks, 256, 0, stream>>>(
        c_feat, g_feat, src_idx, dst_idx, out, E);
}
